// GraphReasoner_77051713290524
// MI455X (gfx1250) — matrix operand order repaired and checked by reading, not compiled, not run on hardware
//
#include <hip/hip_runtime.h>
#include <hip/hip_bf16.h>

// ---------------------------------------------------------------------------
// Graph reasoner, level-scheduled, WMMA f16-in/f32-acc on gfx1250 (wave32).
//   N_NODES=4096, D_MODEL=1024, K=2048, MAX_DEPS=8, N_CLASSES=64
// Working set ~20 MB -> fully L2-resident (192 MB). Critical path = ~70 dep
// levels -> persistent cooperative kernel with grid barriers between levels.
// ---------------------------------------------------------------------------

#define N_NODES   4096
#define D_MODEL   1024
#define MAX_DEPS  8
#define N_CLASSES 64
#define K_DIM     (2 * D_MODEL)   // 2048
#define TILE_M    16
#define NUM_WG    64
#define WG_THREADS 256

typedef _Float16 v16h __attribute__((ext_vector_type(16)));
typedef _Float16 v8h  __attribute__((ext_vector_type(8)));
typedef float    v8f  __attribute__((ext_vector_type(8)));

// ---- workspace layout (bytes) --------------------------------------------
// W1s  : f16 swizzled B fragments [64 kt][64 ct][32 lane][16]  = 4 MB
// ans  : f32 answers [4096][1024]                              = 16 MB
// level/order/start/cursor/meta                                ~ 66 KB
#define WS_W1S_OFF    0ull
#define WS_ANS_OFF    (4ull * 1024 * 1024)
#define WS_LEVEL_OFF  (WS_ANS_OFF + 16ull * 1024 * 1024)
#define WS_ORDER_OFF  (WS_LEVEL_OFF + 4096ull * 4)
#define WS_START_OFF  (WS_ORDER_OFF + 4096ull * 4)
#define WS_CURSOR_OFF (WS_START_OFF + 4104ull * 4)
#define WS_META_OFF   (WS_CURSOR_OFF + 4104ull * 4)
// meta[0] = n_levels, meta[1] = barrier count, meta[2] = barrier generation

// ---------------------------------------------------------------------------
// Kernel 1: swizzle W1 (f32 [1024 out][2048 in]) into per-lane f16 B
// fragments for v_wmma_f32_16x16x32_f16.
// B layout (32x16 KxN tile, wave32): element j of lane L holds
//   K = kt*32 + (L/16)*8 + (j%8) + (j/8)*16 ,  N = ct*16 + (L%16)
// stored contiguously so one 32B load per lane fetches a whole fragment.
// ---------------------------------------------------------------------------
__global__ void k_w1_swizzle(const float* __restrict__ W1,
                             _Float16* __restrict__ W1s) {
  const int total = K_DIM * D_MODEL;  // 2M elements
  for (int o = blockIdx.x * blockDim.x + threadIdx.x; o < total;
       o += gridDim.x * blockDim.x) {
    const int j    = o & 15;
    const int lane = (o >> 4) & 31;
    const int ct   = (o >> 9) & 63;
    const int kt   = o >> 15;
    const int k    = kt * 32 + (lane >> 4) * 8 + (j & 7) + (j >> 3) * 16;
    const int col  = ct * 16 + (lane & 15);
    W1s[o] = (_Float16)W1[(size_t)col * K_DIM + k];
  }
}

// ---------------------------------------------------------------------------
// Kernel 2: topological levels + counting sort (cheap: ~40K scalar ops).
// Also resets the grid-barrier state every call (harness replays the graph
// without re-initializing the workspace).
// ---------------------------------------------------------------------------
__global__ void k_levels(const int* __restrict__ dep_idx,
                         const int* __restrict__ dep_mask,
                         int* __restrict__ level, int* __restrict__ order,
                         int* __restrict__ start, int* __restrict__ cursor,
                         int* __restrict__ meta) {
  if (blockIdx.x != 0 || threadIdx.x != 0) return;
  int nlev = 0;
  for (int i = 0; i < N_NODES; ++i) {
    int lv = 0;
    for (int d = 0; d < MAX_DEPS; ++d) {
      if (dep_mask[i * MAX_DEPS + d]) {
        const int l = level[dep_idx[i * MAX_DEPS + d]] + 1;
        lv = l > lv ? l : lv;
      }
    }
    level[i] = lv;
    if (lv + 1 > nlev) nlev = lv + 1;
  }
  for (int l = 0; l <= nlev; ++l) start[l] = 0;
  for (int i = 0; i < N_NODES; ++i) start[level[i] + 1]++;
  for (int l = 0; l < nlev; ++l) start[l + 1] += start[l];
  for (int l = 0; l <= nlev; ++l) cursor[l] = start[l];
  for (int i = 0; i < N_NODES; ++i) order[cursor[level[i]]++] = i;
  meta[0] = nlev;
  meta[1] = 0;  // barrier arrival count
  meta[2] = 0;  // barrier generation
}

// ---------------------------------------------------------------------------
// Grid-wide barrier (release/acquire atomics + s_sleep backoff).
// ---------------------------------------------------------------------------
__device__ __forceinline__ void grid_barrier(int* cnt, int* gen) {
  __threadfence();          // make this block's global stores device-visible
  __syncthreads();
  if (threadIdx.x == 0) {
    const int g = __hip_atomic_load(gen, __ATOMIC_ACQUIRE,
                                    __HIP_MEMORY_SCOPE_AGENT);
    const int arrived = __hip_atomic_fetch_add(cnt, 1, __ATOMIC_ACQ_REL,
                                               __HIP_MEMORY_SCOPE_AGENT);
    if (arrived == NUM_WG - 1) {
      __hip_atomic_store(cnt, 0, __ATOMIC_RELAXED, __HIP_MEMORY_SCOPE_AGENT);
      __hip_atomic_fetch_add(gen, 1, __ATOMIC_ACQ_REL,
                             __HIP_MEMORY_SCOPE_AGENT);
    } else {
      while (__hip_atomic_load(gen, __ATOMIC_ACQUIRE,
                               __HIP_MEMORY_SCOPE_AGENT) == g) {
        __builtin_amdgcn_s_sleep(1);
      }
    }
  }
  __syncthreads();
}

// XOR-swizzled LDS address (16B-chunk granularity) for the combined buffer:
// keeps A-fragment b128 loads at worst 2-way bank conflicted instead of
// 16-way (row stride 4096B would alias every row onto the same banks).
__device__ __forceinline__ int swz(int m, int k) {
  return (m << 11) + ((((k >> 3) ^ m) << 3) | (k & 7));
}

// ---------------------------------------------------------------------------
// Kernel 3: persistent cooperative forward pass over dependency levels.
// ---------------------------------------------------------------------------
__global__ void __launch_bounds__(WG_THREADS, 1)
k_forward(const float* __restrict__ questions, const float* __restrict__ b1,
          const int* __restrict__ dep_idx, const int* __restrict__ dep_mask,
          const _Float16* __restrict__ W1s, float* __restrict__ answers,
          const int* __restrict__ order, const int* __restrict__ start,
          int* __restrict__ meta) {
  __shared__ _Float16 comb[TILE_M * K_DIM];  // 64 KB, XOR-swizzled

  int* cnt = meta + 1;
  int* gen = meta + 2;
  const int nlev = meta[0];
  const int wave = threadIdx.x >> 5;   // wave32
  const int lane = threadIdx.x & 31;
  const int half = lane >> 4;
  const int nlo  = lane & 15;

  for (int lv = 0; lv < nlev; ++lv) {
    const int s     = start[lv];
    const int cntN  = start[lv + 1] - s;
    const int tiles = (cntN + TILE_M - 1) / TILE_M;

    for (int t = blockIdx.x; t < tiles; t += NUM_WG) {
      // Per-wave copy of this tile's node ids: lane j (j<16) owns row j.
      // Retrieved later with __shfl (ds_bpermute) -> no reloads in epilogue.
      const int myRow  = t * TILE_M + nlo;
      const int myNode = (myRow < cntN) ? order[s + myRow] : -1;

      // ---- stage combined = [q | masked-mean(ctx)] as f16 in LDS ----------
      for (int r = 0; r < 2; ++r) {
        const int m   = wave * 2 + r;
        const int row = t * TILE_M + m;
        if (row < cntN) {
          const int node = order[s + row];
          int di[MAX_DEPS], dm[MAX_DEPS], nd = 0;
#pragma unroll
          for (int d = 0; d < MAX_DEPS; ++d) {
            di[d] = dep_idx[node * MAX_DEPS + d];
            dm[d] = dep_mask[node * MAX_DEPS + d];
            nd += dm[d];
          }
          const float inv = 1.0f / fmaxf((float)nd, 1.0f);
          const float4* qrow = (const float4*)(questions + (size_t)node * D_MODEL);
          for (int c4 = lane; c4 < D_MODEL / 4; c4 += 32) {
            const float4 qv = qrow[c4];
            float ax = 0.f, ay = 0.f, az = 0.f, aw = 0.f;
#pragma unroll
            for (int d = 0; d < MAX_DEPS; ++d) {
              if (dm[d]) {
                const float4 v =
                    ((const float4*)(answers + (size_t)di[d] * D_MODEL))[c4];
                ax += v.x; ay += v.y; az += v.z; aw += v.w;
              }
            }
            const int c = c4 * 4;
            comb[swz(m, c + 0)] = (_Float16)qv.x;
            comb[swz(m, c + 1)] = (_Float16)qv.y;
            comb[swz(m, c + 2)] = (_Float16)qv.z;
            comb[swz(m, c + 3)] = (_Float16)qv.w;
            comb[swz(m, D_MODEL + c + 0)] = (_Float16)(ax * inv);
            comb[swz(m, D_MODEL + c + 1)] = (_Float16)(ay * inv);
            comb[swz(m, D_MODEL + c + 2)] = (_Float16)(az * inv);
            comb[swz(m, D_MODEL + c + 3)] = (_Float16)(aw * inv);
          }
        } else {
          for (int k = lane; k < K_DIM; k += 32)
            comb[(m << 11) + k] = (_Float16)0.f;
        }
      }
      __syncthreads();

      // ---- GEMM: each wave owns 8 column tiles (128 of 1024 outputs) ------
      for (int cti = 0; cti < 8; ++cti) {
        const int ct = wave * 8 + cti;
        const float bias = b1[ct * 16 + nlo];  // C/D: N = lane%16
        v8f acc;
#pragma unroll
        for (int r = 0; r < 8; ++r) acc[r] = bias;

#pragma unroll 4
        for (int kt = 0; kt < K_DIM / 32; ++kt) {
          // A fragment (16x32 f16): lo = K[half*8 .. +8), hi = K[16+half*8 ..)
          const int kb = kt * 32 + half * 8;
          const v8h lo = *(const v8h*)&comb[swz(nlo, kb)];
          const v8h hi = *(const v8h*)&comb[swz(nlo, kb + 16)];
          v16h a;
#pragma unroll
          for (int j = 0; j < 8; ++j) { a[j] = lo[j]; a[8 + j] = hi[j]; }

          // B fragment: one contiguous 32B load per lane (pre-swizzled)
          const _Float16* bp =
              W1s + ((size_t)(kt * 64 + ct) * 32 + lane) * 16;
          const v16h b = *(const v16h*)bp;

          // WGP-scope prefetch of the K-tile 4 steps ahead (locality 3 ->
          // pull into all cache levels; W1s is L2-resident, this hides the
          // L2->WGP hop). One prefetch per 4 WMMAs.
          if ((kt & 3) == 0 && kt + 4 < K_DIM / 32)
            __builtin_prefetch(bp + 4 * 64 * 32 * 16, 0, 3);

          acc = __builtin_amdgcn_wmma_f32_16x16x32_f16(
              false, a, false, b, (short)0, acc, false, false);
        }

        // ---- exact GELU + scatter store (C/D: M = r + 8*half, N = nlo) ----
#pragma unroll
        for (int r = 0; r < 8; ++r) {
          const int M    = half * 8 + r;
          const int node = __shfl(myNode, M, 32);  // row t*16+M's node id
          if (node >= 0) {
            const float x = acc[r];
            const float g = 0.5f * x * (1.0f + erff(x * 0.70710678118f));
            answers[(size_t)node * D_MODEL + ct * 16 + nlo] = g;
          }
        }
      }
      __syncthreads();  // comb reused by next tile
    }
    grid_barrier(cnt, gen);  // level lv answers visible before level lv+1
  }
}

// ---------------------------------------------------------------------------
// Kernel 4: logits = answers[4095] @ W2^T + b2   (64 x 1024 dot products)
// ---------------------------------------------------------------------------
__global__ void k_logits(const float* __restrict__ answers,
                         const float* __restrict__ W2,
                         const float* __restrict__ b2,
                         float* __restrict__ out) {
  const int c = blockIdx.x * blockDim.x + threadIdx.x;
  if (c < N_CLASSES) {
    const float* root = answers + (size_t)(N_NODES - 1) * D_MODEL;
    float acc = b2[c];
    for (int d = 0; d < D_MODEL; ++d)
      acc += root[d] * W2[(size_t)c * D_MODEL + d];
    out[c] = acc;
  }
}

// ---------------------------------------------------------------------------
extern "C" void kernel_launch(void* const* d_in, const int* in_sizes, int n_in,
                              void* d_out, int out_size, void* d_ws,
                              size_t ws_size, hipStream_t stream) {
  (void)in_sizes; (void)n_in; (void)out_size; (void)ws_size;  // ~20.1 MB used
  const float* questions = (const float*)d_in[0];
  const float* W1        = (const float*)d_in[1];
  const float* b1        = (const float*)d_in[2];
  const float* W2        = (const float*)d_in[3];
  const float* b2        = (const float*)d_in[4];
  const int*   dep_idx   = (const int*)d_in[5];
  const int*   dep_mask  = (const int*)d_in[6];

  char* ws = (char*)d_ws;
  _Float16* W1s   = (_Float16*)(ws + WS_W1S_OFF);
  float*    ans   = (float*)(ws + WS_ANS_OFF);
  int*      level = (int*)(ws + WS_LEVEL_OFF);
  int*      order = (int*)(ws + WS_ORDER_OFF);
  int*      startp= (int*)(ws + WS_START_OFF);
  int*      curs  = (int*)(ws + WS_CURSOR_OFF);
  int*      meta  = (int*)(ws + WS_META_OFF);

  k_w1_swizzle<<<512, 256, 0, stream>>>(W1, W1s);
  k_levels<<<1, 1, 0, stream>>>(dep_idx, dep_mask, level, order, startp, curs,
                                meta);
  k_forward<<<NUM_WG, WG_THREADS, 0, stream>>>(questions, b1, dep_idx,
                                               dep_mask, W1s, ans, order,
                                               startp, meta);
  k_logits<<<1, 64, 0, stream>>>(ans, W2, b2, (float*)d_out);
}
